// VAE_38242388804192
// MI455X (gfx1250) — matrix operand order repaired and checked by reading, not compiled, not run on hardware
//
#include <hip/hip_runtime.h>
#include <hip/hip_bf16.h>
#include <stdint.h>

typedef _Float16 half_t;
typedef __attribute__((ext_vector_type(8)))  _Float16 v8h;
typedef __attribute__((ext_vector_type(16))) _Float16 v16h;
typedef __attribute__((ext_vector_type(8)))  float    v8f;

#define SLOPE 0.01f
#define RECON_ELEMS (256 * 3 * 32 * 32)

// ------------------------------------------------------------------
// Weight repack: conv/convT weights -> f16 [Cout][Kpad], K-order (kh,kw,ci)
// transposed=1: src layout [Cin][Cout][KH][KW] with spatial flip (convT)
// ------------------------------------------------------------------
__global__ void repack_conv_w(const float* __restrict__ w, half_t* __restrict__ wp,
                              int Cout, int Cin, int KH, int KW, int Kpad, int transposed)
{
    int i = blockIdx.x * blockDim.x + threadIdx.x;
    int total = Cout * Kpad;
    if (i >= total) return;
    int n = i / Kpad, k = i - n * Kpad;
    int K = Cin * KH * KW;
    float v = 0.f;
    if (k < K) {
        int ci = k % Cin; int r = k / Cin; int kw = r % KW; int kh = r / KW;
        if (!transposed)
            v = w[(((size_t)n * Cin + ci) * KH + kh) * KW + kw];
        else
            v = w[(((size_t)ci * Cout + n) * KH + (KH - 1 - kh)) * KW + (KW - 1 - kw)];
    }
    wp[i] = (half_t)v;
}

// Dense weight repack -> f16 [N][Kpad].
// kperm=1: e_wout, map dst k=(h*4+w)*256+c -> src k=c*16+(h*4+w)   (K=4096)
// nperm=1: d_w2,  map dst n=(h*8+w)*64+c  -> src n=c*64+(h*8+w)   (N=4096)
__global__ void repack_dense_w(const float* __restrict__ w, half_t* __restrict__ wp,
                               int N, int K, int Kpad, int kperm, int nperm)
{
    int i = blockIdx.x * blockDim.x + threadIdx.x;
    int total = N * Kpad;
    if (i >= total) return;
    int n = i / Kpad, k = i - n * Kpad;
    float v = 0.f;
    if (k < K) {
        int ksrc = k, nsrc = n;
        if (kperm) { int c = k & 255; int s = k >> 8; ksrc = c * 16 + s; }
        if (nperm) { int c = n & 63;  int s = n >> 6; nsrc = c * 64 + s; }
        v = w[(size_t)nsrc * K + ksrc];
    }
    wp[i] = (half_t)v;
}

// ------------------------------------------------------------------
// A-fragment loader: 16x32 f16 tile row for one lane from a contiguous
// 32-half chunk at p (ISA layout: halves 0..7 -> K offs g*8..g*8+7,
// halves 8..15 -> K offs 16+g*8..16+g*8+7). Two 16B vector loads.
// ------------------------------------------------------------------
__device__ __forceinline__ v16h load_a32(const half_t* p, int g, bool valid)
{
    v8h lo = {(half_t)0, (half_t)0, (half_t)0, (half_t)0,
              (half_t)0, (half_t)0, (half_t)0, (half_t)0};
    v8h hi = lo;
    if (valid) {
        lo = *(const v8h*)(p + g * 8);
        hi = *(const v8h*)(p + 16 + g * 8);
    }
    return __builtin_shufflevector(lo, hi, 0, 1, 2, 3, 4, 5, 6, 7,
                                   8, 9, 10, 11, 12, 13, 14, 15);
}

// 4 WMMA ops sharing one A fragment (16x64 output per wave).
__device__ __forceinline__ void gemm_step(v8f acc[4], v16h af,
                                          const half_t* const* wrow, int k0)
{
#pragma unroll
    for (int s = 0; s < 4; ++s) {
        v8h blo = *(const v8h*)(wrow[s] + k0);
        v8h bhi = *(const v8h*)(wrow[s] + k0 + 16);
        v16h bf = __builtin_shufflevector(blo, bhi, 0, 1, 2, 3, 4, 5, 6, 7,
                                          8, 9, 10, 11, 12, 13, 14, 15);
        acc[s] = __builtin_amdgcn_wmma_f32_16x16x32_f16(false, af, false, bf,
                                                        (short)0, acc[s], false, false);
    }
}

// ------------------------------------------------------------------
// Implicit-GEMM: C[M,N] = act( A_gather[M,K] * Wp^T + bias ), f16 out.
// Wp f16 [N][Kpad]. Wave computes 16(M) x 64(N).
// MODE 0: dense f16 A [M,lda], K multiple of 32, A padded to lda.
// MODE 1: conv gather NHWC f16, Cin % 32 == 0 (vectorized taps).
// MODE 2: conv gather NCHW f32, generic scalar (first layer only).
// ------------------------------------------------------------------
template <int MODE>
__global__ void wmma_gemm(const void* __restrict__ Asrc, const half_t* __restrict__ Wp,
                          const float* __restrict__ bias, half_t* __restrict__ Cdst,
                          int M, int N, int K, int Kpad, int lda, int act,
                          int Cin, int H, int W, int KH, int KW,
                          int ostride, int pad, int idil, int OH, int OW)
{
    int wave = threadIdx.x >> 5;
    int lane = threadIdx.x & 31;
    int tiles_n4 = (N + 63) >> 6;
    int tiles_m  = M >> 4;                    // all layer M's are multiples of 16
    int tile = blockIdx.x * (blockDim.x >> 5) + wave;
    if (tile >= tiles_m * tiles_n4) return;   // wave-uniform exit: EXEC all-ones at WMMA
    int tm = tile / tiles_n4, tn = tile - tm * tiles_n4;

    int g  = lane >> 4;
    int mr = lane & 15;
    int m  = tm * 16 + mr;

    int nn[4];
    const half_t* wrow[4];
#pragma unroll
    for (int s = 0; s < 4; ++s) {
        nn[s]   = tn * 64 + s * 16 + mr;
        wrow[s] = Wp + (size_t)nn[s] * Kpad + g * 8;  // rows beyond N read padded alloc
    }

    v8f acc[4];
#pragma unroll
    for (int s = 0; s < 4; ++s)
#pragma unroll
        for (int r = 0; r < 8; ++r) acc[s][r] = 0.f;

    if (MODE == 0) {
        const half_t* arow = (const half_t*)Asrc + (size_t)m * lda;
        for (int k0 = 0; k0 < K; k0 += 32) {
            __builtin_prefetch((const void*)(wrow[0] + k0 + 128), 0, 1);
            v16h af = load_a32(arow + k0, g, true);
            gemm_step(acc, af, wrow, k0);
        }
    } else if (MODE == 1) {
        int s0 = m; int ow = s0 % OW; s0 /= OW; int oh = s0 % OH; int b = s0 / OH;
        const half_t* abase = (const half_t*)Asrc;
        int k0 = 0;
        for (int kh = 0; kh < KH; ++kh) {
            int ih = oh * ostride - pad + kh;
            bool vh = true;
            if (idil != 1) { if ((ih % idil) == 0) ih /= idil; else vh = false; }
            vh = vh && (ih >= 0) && (ih < H);
            for (int kw = 0; kw < KW; ++kw) {
                int iw = ow * ostride - pad + kw;
                bool v = vh;
                if (idil != 1) { if ((iw % idil) == 0) iw /= idil; else v = false; }
                v = v && (iw >= 0) && (iw < W);
                const half_t* p = abase + (((size_t)b * H + ih) * W + iw) * Cin;
                for (int ci0 = 0; ci0 < Cin; ci0 += 32, k0 += 32) {
                    __builtin_prefetch((const void*)(wrow[0] + k0 + 128), 0, 1);
                    v16h af = load_a32(p + ci0, g, v);
                    gemm_step(acc, af, wrow, k0);
                }
            }
        }
    } else {  // MODE 2: generic scalar gather, NCHW f32 (enc layer 1, K=27)
        int s0 = m; int ow = s0 % OW; s0 /= OW; int oh = s0 % OH; int b = s0 / OH;
        for (int k0 = 0; k0 < K; k0 += 32) {
            v16h af;
#pragma unroll
            for (int h = 0; h < 16; ++h) {
                int k = k0 + ((h < 8) ? (g * 8 + h) : (16 + g * 8 + (h - 8)));
                half_t val = (half_t)0.f;
                if (k < K) {
                    int ci = k % Cin; int r = k / Cin;
                    int kw = r % KW;  int kh = r / KW;
                    int ih = oh * ostride - pad + kh;
                    int iw = ow * ostride - pad + kw;
                    if (ih >= 0 && ih < H && iw >= 0 && iw < W)
                        val = (half_t)(((const float*)Asrc)[(((size_t)b * Cin + ci) * H + ih) * W + iw]);
                }
                af[h] = val;
            }
            gemm_step(acc, af, wrow, k0);
        }
    }

    // epilogue: bias + optional leaky-relu, store f16 [M,N]
#pragma unroll
    for (int s = 0; s < 4; ++s) {
        if (nn[s] < N) {
            float bv = bias ? bias[nn[s]] : 0.f;
#pragma unroll
            for (int r = 0; r < 8; ++r) {
                float v = acc[s][r] + bv;
                if (act) v = (v > 0.f) ? v : SLOPE * v;
                Cdst[(size_t)(tm * 16 + g * 8 + r) * N + nn[s]] = (half_t)v;
            }
        }
    }
}

// ------------------------------------------------------------------
// BatchNorm (training-mode batch stats) over NHWC columns.
// ------------------------------------------------------------------
__global__ void bn_stats(const half_t* __restrict__ a, const float* __restrict__ gamma,
                         const float* __restrict__ beta, float* __restrict__ ss,
                         int Mrows, int N)
{
    int c = blockIdx.x;
    __shared__ float s0[256], s1[256];
    float s = 0.f, q = 0.f;
    for (int i = threadIdx.x; i < Mrows; i += blockDim.x) {
        float v = (float)a[(size_t)i * N + c];
        s += v; q += v * v;
    }
    s0[threadIdx.x] = s; s1[threadIdx.x] = q;
    __syncthreads();
    for (int st = 128; st > 0; st >>= 1) {
        if (threadIdx.x < st) { s0[threadIdx.x] += s0[threadIdx.x + st];
                                s1[threadIdx.x] += s1[threadIdx.x + st]; }
        __syncthreads();
    }
    if (threadIdx.x == 0) {
        float mean = s0[0] / (float)Mrows;
        float var  = s1[0] / (float)Mrows - mean * mean;
        float inv  = rsqrtf(var + 1e-5f);
        float sc   = gamma[c] * inv;
        ss[2 * c]     = sc;
        ss[2 * c + 1] = beta[c] - mean * sc;
    }
}

__global__ void bn_apply_lrelu(half_t* __restrict__ a, const float* __restrict__ ss,
                               size_t total, int N)
{
    size_t i = (size_t)blockIdx.x * blockDim.x + threadIdx.x;
    if (i >= total) return;
    int c = (int)(i % N);
    float v = (float)a[i] * ss[2 * c] + ss[2 * c + 1];
    a[i] = (half_t)((v > 0.f) ? v : SLOPE * v);
}

// ------------------------------------------------------------------
// dist[b,j] = || codes[b,j,:] - emb[b,:] ||^2     (one wave per j)
// ------------------------------------------------------------------
__global__ void dist_kernel(const half_t* __restrict__ emb, const float* __restrict__ codes,
                            float* __restrict__ dist)
{
    int b = blockIdx.x;
    __shared__ float e[128];
    if (threadIdx.x < 128) e[threadIdx.x] = (float)emb[b * 128 + threadIdx.x];
    __syncthreads();
    int wave = threadIdx.x >> 5, lane = threadIdx.x & 31;
    int nwaves = blockDim.x >> 5;
    for (int j = wave; j < 1024; j += nwaves) {
        const float* cp = codes + ((size_t)b * 1024 + j) * 128;
        float s = 0.f;
        for (int l = lane; l < 128; l += 32) { float d = cp[l] - e[l]; s += d * d; }
        for (int off = 16; off > 0; off >>= 1) s += __shfl_down(s, off, 32);
        if (lane == 0) dist[(size_t)b * 1024 + j] = s;
    }
}

// Greedy exclusive argmin scan (serial over 256 rows; one 1024-thread block).
__global__ void greedy_assign(const float* __restrict__ dist, float* __restrict__ out)
{
    __shared__ float sval[1024];
    __shared__ int   sidx[1024];
    __shared__ int   taken[1024];
    int t = threadIdx.x;
    taken[t] = 0;
    __syncthreads();
    float sum = 0.f;
    for (int b = 0; b < 256; ++b) {
        sval[t] = taken[t] ? 99999.0f : dist[(size_t)b * 1024 + t];
        sidx[t] = t;
        __syncthreads();
        for (int s = 512; s > 0; s >>= 1) {
            if (t < s) {
                float v2 = sval[t + s]; int i2 = sidx[t + s];
                if (v2 < sval[t] || (v2 == sval[t] && i2 < sidx[t])) { sval[t] = v2; sidx[t] = i2; }
            }
            __syncthreads();
        }
        if (t == 0) {
            sum += sval[0];
            taken[sidx[0]] = 1;
            out[RECON_ELEMS + 1 + b] = (float)sidx[0];
        }
        __syncthreads();
    }
    if (t == 0) out[RECON_ELEMS] = sum;
}

// z = concat(emb, task_bits, zero-pad) as f16 [256,160] (lda padded for vector loads)
__global__ void zcat_kernel(const half_t* __restrict__ emb, const int* __restrict__ task_id,
                            half_t* __restrict__ zc)
{
    int i = blockIdx.x * blockDim.x + threadIdx.x;
    if (i >= 256 * 160) return;
    int b = i / 160, c = i - b * 160;
    half_t v = (half_t)0.f;
    if (c < 128) v = emb[b * 128 + c];
    else if (c < 136) {
        int code = (task_id[0] * 13) & 255;
        v = (half_t)(float)((code >> (7 - (c - 128))) & 1);
    }
    zc[i] = v;
}

// dec4 NHWC f16 [B,32,32,3] -> d_out NCHW f32 [B,3,32,32]
__global__ void finalize_recon(const half_t* __restrict__ dec4, float* __restrict__ out)
{
    int i = blockIdx.x * blockDim.x + threadIdx.x;
    if (i >= RECON_ELEMS) return;
    int w = i & 31; int r = i >> 5;
    int h = r & 31; r >>= 5;
    int c = r % 3;  int b = r / 3;
    out[i] = (float)dec4[(((size_t)b * 32 + h) * 32 + w) * 3 + c];
}

// ------------------------------------------------------------------
extern "C" void kernel_launch(void* const* d_in, const int* in_sizes, int n_in,
                              void* d_out, int out_size, void* d_ws, size_t ws_size,
                              hipStream_t stream)
{
    const float* x       = (const float*)d_in[0];
    const int*   task_id = (const int*)  d_in[1];
    const float* codes   = (const float*)d_in[2];
    const float* e_w1  = (const float*)d_in[3];
    const float* e_g1  = (const float*)d_in[4];
    const float* e_b1  = (const float*)d_in[5];
    const float* e_w2  = (const float*)d_in[6];
    const float* e_g2  = (const float*)d_in[7];
    const float* e_b2  = (const float*)d_in[8];
    const float* e_w3  = (const float*)d_in[9];
    const float* e_g3  = (const float*)d_in[10];
    const float* e_b3  = (const float*)d_in[11];
    const float* e_wout= (const float*)d_in[12];
    const float* e_bout= (const float*)d_in[13];
    const float* d_w1  = (const float*)d_in[14];
    const float* d_b1  = (const float*)d_in[15];
    const float* d_w2  = (const float*)d_in[16];
    const float* d_b2  = (const float*)d_in[17];
    const float* dc1_w = (const float*)d_in[18];
    const float* dc1_g = (const float*)d_in[19];
    const float* dc1_b = (const float*)d_in[20];
    const float* dc2_w = (const float*)d_in[21];
    const float* dc2_g = (const float*)d_in[22];
    const float* dc2_b = (const float*)d_in[23];
    const float* dc3_w = (const float*)d_in[24];
    const float* dc3_g = (const float*)d_in[25];
    const float* dc3_b = (const float*)d_in[26];
    const float* dc4_w = (const float*)d_in[27];
    (void)in_sizes; (void)n_in; (void)out_size; (void)ws_size;

    size_t off = 0;
    auto alloc = [&](size_t bytes) -> void* {
        void* p = (char*)d_ws + off;
        off += (bytes + 255) & ~(size_t)255;
        return p;
    };
    // packed f16 weights [N][Kpad]  (wp_c4 padded to 16 rows for OOB-safe B reads)
    half_t* wp_e1  = (half_t*)alloc(sizeof(half_t) * (size_t)64   * 32);
    half_t* wp_e2  = (half_t*)alloc(sizeof(half_t) * (size_t)128  * 576);
    half_t* wp_e3  = (half_t*)alloc(sizeof(half_t) * (size_t)256  * 1152);
    half_t* wp_eo  = (half_t*)alloc(sizeof(half_t) * (size_t)128  * 4096);
    half_t* wp_d1  = (half_t*)alloc(sizeof(half_t) * (size_t)192  * 160);
    half_t* wp_d2  = (half_t*)alloc(sizeof(half_t) * (size_t)4096 * 192);
    half_t* wp_c1  = (half_t*)alloc(sizeof(half_t) * (size_t)256  * 576);
    half_t* wp_c2  = (half_t*)alloc(sizeof(half_t) * (size_t)384  * 4096);
    half_t* wp_c3  = (half_t*)alloc(sizeof(half_t) * (size_t)320  * 6144);
    half_t* wp_c4  = (half_t*)alloc(sizeof(half_t) * (size_t)16   * 5120);
    // activations (NHWC f16) + small fp32 scratch
    half_t* act1 = (half_t*)alloc(sizeof(half_t) * (size_t)65536  * 64);
    half_t* act2 = (half_t*)alloc(sizeof(half_t) * (size_t)16384  * 128);
    half_t* act3 = (half_t*)alloc(sizeof(half_t) * (size_t)4096   * 256);
    half_t* emb  = (half_t*)alloc(sizeof(half_t) * (size_t)256    * 128);
    half_t* zc   = (half_t*)alloc(sizeof(half_t) * (size_t)256    * 160);
    half_t* z1   = (half_t*)alloc(sizeof(half_t) * (size_t)256    * 192);
    half_t* z2   = (half_t*)alloc(sizeof(half_t) * (size_t)256    * 4096);
    half_t* dec1 = (half_t*)alloc(sizeof(half_t) * (size_t)43264  * 256);
    half_t* dec2 = (half_t*)alloc(sizeof(half_t) * (size_t)200704 * 384);
    half_t* dec3 = (half_t*)alloc(sizeof(half_t) * (size_t)246016 * 320);
    half_t* dec4 = (half_t*)alloc(sizeof(half_t) * (size_t)262144 * 3);
    float*  dist = (float*)alloc(sizeof(float)  * (size_t)256 * 1024);
    float*  bnss = (float*)alloc(sizeof(float)  * 2 * 512);

    auto rpc = [&](const float* w, half_t* wp, int Cout, int Cin, int KH, int KW, int Kpad, int tr) {
        int total = Cout * Kpad;
        repack_conv_w<<<(total + 255) / 256, 256, 0, stream>>>(w, wp, Cout, Cin, KH, KW, Kpad, tr);
    };
    auto rpd = [&](const float* w, half_t* wp, int N, int K, int Kpad, int kperm, int nperm) {
        int total = N * Kpad;
        repack_dense_w<<<(total + 255) / 256, 256, 0, stream>>>(w, wp, N, K, Kpad, kperm, nperm);
    };
    auto gemm = [&](int mode, const void* A, const half_t* Wp, const float* bias, half_t* C,
                    int M, int N, int K, int Kpad, int lda, int act,
                    int Cin, int H, int W, int KH, int KW, int os, int pad, int idil,
                    int OH, int OW) {
        int tiles  = (M >> 4) * ((N + 63) >> 6);
        int blocks = (tiles + 7) / 8;                  // 8 waves (256 thr) per block
        if (mode == 0)
            wmma_gemm<0><<<blocks, 256, 0, stream>>>(A, Wp, bias, C, M, N, K, Kpad, lda, act,
                                                     Cin, H, W, KH, KW, os, pad, idil, OH, OW);
        else if (mode == 1)
            wmma_gemm<1><<<blocks, 256, 0, stream>>>(A, Wp, bias, C, M, N, K, Kpad, lda, act,
                                                     Cin, H, W, KH, KW, os, pad, idil, OH, OW);
        else
            wmma_gemm<2><<<blocks, 256, 0, stream>>>(A, Wp, bias, C, M, N, K, Kpad, lda, act,
                                                     Cin, H, W, KH, KW, os, pad, idil, OH, OW);
    };
    auto bn = [&](half_t* a, const float* g, const float* b, int M, int N) {
        bn_stats<<<N, 256, 0, stream>>>(a, g, b, bnss, M, N);
        size_t total = (size_t)M * N;
        bn_apply_lrelu<<<(unsigned)((total + 255) / 256), 256, 0, stream>>>(a, bnss, total, N);
    };

    // ---- weight repack ----
    rpc(e_w1,  wp_e1, 64,  3,   3, 3, 32,   0);
    rpc(e_w2,  wp_e2, 128, 64,  3, 3, 576,  0);
    rpc(e_w3,  wp_e3, 256, 128, 3, 3, 1152, 0);
    rpd(e_wout,wp_eo, 128, 4096, 4096, 1, 0);
    rpd(d_w1,  wp_d1, 192, 136,  160,  0, 0);
    rpd(d_w2,  wp_d2, 4096, 192, 192,  0, 1);
    rpc(dc1_w, wp_c1, 256, 64,  3, 3, 576,  1);
    rpc(dc2_w, wp_c2, 384, 256, 4, 4, 4096, 1);
    rpc(dc3_w, wp_c3, 320, 384, 4, 4, 6144, 1);
    rpc(dc4_w, wp_c4, 3,   320, 4, 4, 5120, 1);

    // ---- encoder ----
    gemm(2, x,    wp_e1, nullptr, act1, 65536, 64,  27,   32,   0, 0, 3,   32, 32, 3, 3, 2, 1, 1, 16, 16);
    bn(act1, e_g1, e_b1, 65536, 64);
    gemm(1, act1, wp_e2, nullptr, act2, 16384, 128, 576,  576,  0, 0, 64,  16, 16, 3, 3, 2, 1, 1, 8, 8);
    bn(act2, e_g2, e_b2, 16384, 128);
    gemm(1, act2, wp_e3, nullptr, act3, 4096,  256, 1152, 1152, 0, 0, 128, 8,  8,  3, 3, 2, 1, 1, 4, 4);
    bn(act3, e_g3, e_b3, 4096, 256);
    gemm(0, act3, wp_eo, e_bout, emb, 256, 128, 4096, 4096, 4096, 0, 0,0,0,0,0,1,0,1,0,0);

    // ---- code assignment ----
    dist_kernel<<<256, 256, 0, stream>>>(emb, codes, dist);
    greedy_assign<<<1, 1024, 0, stream>>>(dist, (float*)d_out);

    // ---- decoder ----
    zcat_kernel<<<(256 * 160 + 255) / 256, 256, 0, stream>>>(emb, task_id, zc);
    gemm(0, zc, wp_d1, d_b1, z1, 256, 192,  160, 160, 160, 1, 0,0,0,0,0,1,0,1,0,0);
    gemm(0, z1, wp_d2, d_b2, z2, 256, 4096, 192, 192, 192, 1, 0,0,0,0,0,1,0,1,0,0);
    // z2 = NHWC seed [256,8,8,64]
    gemm(1, z2,   wp_c1, nullptr, dec1, 43264,  256, 576,  576,  0, 0, 64,  8,  8,  3, 3, 1, 0, 2, 13, 13);
    bn(dec1, dc1_g, dc1_b, 43264, 256);
    gemm(1, dec1, wp_c2, nullptr, dec2, 200704, 384, 4096, 4096, 0, 0, 256, 13, 13, 4, 4, 1, 3, 2, 28, 28);
    bn(dec2, dc2_g, dc2_b, 200704, 384);
    gemm(1, dec2, wp_c3, nullptr, dec3, 246016, 320, 6144, 6144, 0, 0, 384, 28, 28, 4, 4, 1, 3, 1, 31, 31);
    bn(dec3, dc3_g, dc3_b, 246016, 320);
    gemm(1, dec3, wp_c4, nullptr, dec4, 262144, 3,   5120, 5120, 0, 0, 320, 31, 31, 4, 4, 1, 2, 1, 32, 32);

    finalize_recon<<<(RECON_ELEMS + 255) / 256, 256, 0, stream>>>(dec4, (float*)d_out);
}
